// PRNNeck_53154515255409
// MI455X (gfx1250) — compile-verified, matrix operand order checked by reading, not run on hardware
//
#include <hip/hip_runtime.h>
#include <hip/hip_bf16.h>
#include <math.h>

typedef _Float16 half_t;
typedef __attribute__((ext_vector_type(16))) _Float16 v16h;
typedef __attribute__((ext_vector_type(8)))  float    v8f;

#define KPT   14
#define CDIM  64
#define NB    16
#define SEQ   2062      /* MAX_LEN + K */

// ---------------------------------------------------------------------------
// WMMA fragment helpers (layouts per CDNA5 ISA §7.12.2, wave32).
// 16-bit A 16x32 / B 32x16 lane striping folds to two contiguous 8-elem runs:
//   element e in [0,8):  K = 8*hf + e
//   element e in [8,16): K = 16 + 8*hf + (e-8)
// f32 C/D 16x16: lane L holds N=L%16; VGPR r holds row M = r + 8*(L/16).
// ---------------------------------------------------------------------------
__device__ __forceinline__ v16h frag_f32(const float* p, int hf) {
  v16h a;
  const float* p0 = p + 8 * hf;
  const float* p1 = p + 16 + 8 * hf;
#pragma unroll
  for (int e = 0; e < 8; ++e) a[e] = (half_t)p0[e];
#pragma unroll
  for (int e = 0; e < 8; ++e) a[e + 8] = (half_t)p1[e];
  return a;
}

__device__ __forceinline__ v16h frag_f16(const half_t* p, int hf) {
  v16h a;
  const half_t* p0 = p + 8 * hf;
  const half_t* p1 = p + 16 + 8 * hf;
#pragma unroll
  for (int e = 0; e < 8; ++e) a[e] = p0[e];
#pragma unroll
  for (int e = 0; e < 8; ++e) a[e + 8] = p1[e];
  return a;
}

// K = 16 (head dim) zero-padded to 32: upper 8 elements compile-time zero.
__device__ __forceinline__ v16h frag_f16_pad16(const half_t* p, int hf) {
  v16h a = {};
  const half_t* p0 = p + 8 * hf;
#pragma unroll
  for (int e = 0; e < 8; ++e) a[e] = p0[e];
  return a;
}

__device__ __forceinline__ v8f wmma16(v16h a, v16h b, v8f c) {
  return __builtin_amdgcn_wmma_f32_16x16x32_f16(false, a, false, b, (short)0, c,
                                                false, false);
}

// wave-local LDS ordering: DS ops are in-order per wave; s_wait_dscnt 0
// guarantees prior LDS writes have landed before subsequent LDS reads.
__device__ __forceinline__ void lds_fence() {
  asm volatile("s_wait_dscnt 0" ::: "memory");
}

__device__ __forceinline__ int seg_start(const int* offset, int b) {
  return b ? offset[b - 1] : 0;
}

__device__ __forceinline__ int imin(int a, int b) { return a < b ? a : b; }

// LayerNorm of (xr[i] + add[i]) over 64 channels, written back to xr.
__device__ __forceinline__ void ln_row64(float* xr, const float* add,
                                         const float* g, const float* b) {
  float s = 0.f;
  for (int i = 0; i < 64; ++i) s += xr[i] + add[i];
  float mu = s * (1.f / 64.f);
  float var = 0.f;
  for (int i = 0; i < 64; ++i) {
    float d = xr[i] + add[i] - mu;
    var += d * d;
  }
  var *= (1.f / 64.f);
  float inv = rsqrtf(var + 1e-5f);
  for (int i = 0; i < 64; ++i)
    xr[i] = (xr[i] + add[i] - mu) * inv * g[i] + b[i];
}

// ---------------------------------------------------------------------------
// 0. zero accumulators
// ---------------------------------------------------------------------------
__global__ void init_kernel(float* num, float* den) {
  int i = blockIdx.x * blockDim.x + threadIdx.x;
  if (i < NB * 15 * 3) num[i] = 0.f;
  if (i < NB * 15) den[i] = 0.f;
}

// ---------------------------------------------------------------------------
// 1. per-batch segment max / mean over feat (N x 64)
// ---------------------------------------------------------------------------
__global__ __launch_bounds__(64) void pool_kernel(const float* feat,
                                                  const int* offset,
                                                  float* fmax, float* fmean) {
  int b = blockIdx.x, c = threadIdx.x;
  int s = seg_start(offset, b), e = offset[b];
  float mx = -3.4e38f, sm = 0.f;
  for (int i = s; i < e; ++i) {
    float v = feat[(size_t)i * CDIM + c];
    mx = fmaxf(mx, v);
    sm += v;
  }
  fmax[b * CDIM + c] = mx;
  fmean[b * CDIM + c] = sm / (float)(e - s);
}

// ---------------------------------------------------------------------------
// 2. seg/reg MLP heads via WMMA + softmax + masked pose pooling
// ---------------------------------------------------------------------------
struct HeadParams {
  const float *w1, *b1, *g1, *bb1, *w2, *b2, *g2, *bb2, *w3, *b3;
};

__global__ __launch_bounds__(128) void heads_kernel(
    const float* feat, const float* coord, const int* offset,
    const float* fmax, const float* fmean, HeadParams segp, HeadParams regp,
    float* seg_out, float* num, float* den, int N) {
  __shared__ half_t h1s[4][16 * 128];
  __shared__ half_t h2s[4][16 * 64];
  __shared__ float  outs[4][2][16 * 16];

  const int wave = threadIdx.x >> 5, lane = threadIdx.x & 31;
  const int lm = lane & 15, hf = lane >> 4;
  const int row0 = (blockIdx.x * 4 + wave) * 16;
  if (row0 >= N) return;

  const int rowC = imin(row0 + lm, N - 1);   // clamped A-row for this lane
  int bidA = 0;
  for (int i = 0; i < NB; ++i) bidA += (rowC >= offset[i]) ? 1 : 0;
  bidA = imin(bidA, NB - 1);

  const float bnscale = rsqrtf(1.f + 1e-5f);
  HeadParams hp2[2] = {segp, regp};

  // precompute the three feat_all source rows (clamped -> always valid)
  const float* arow_feat = feat + (size_t)rowC * 64;
  const float* arow_max = fmax + bidA * 64;
  const float* arow_mean = fmean + bidA * 64;

  for (int hi = 0; hi < 2; ++hi) {
    HeadParams hp = hp2[hi];
    // ---- layer 1: 192 -> 128, bn, relu ----
    for (int nt = 0; nt < 8; ++nt) {
      v8f acc = {};
#pragma unroll
      for (int kt = 0; kt < 6; ++kt) {
        const float* ap = (kt < 2)   ? arow_feat + kt * 32
                          : (kt < 4) ? arow_max + (kt - 2) * 32
                                     : arow_mean + (kt - 4) * 32;
        v16h a = frag_f32(ap, hf);
        v16h w = frag_f32(hp.w1 + (nt * 16 + lm) * 192 + kt * 32, hf);
        acc = wmma16(a, w, acc);
      }
      int och = nt * 16 + lm;
      float sc = hp.g1[och] * bnscale, of = hp.bb1[och], bs = hp.b1[och];
#pragma unroll
      for (int r = 0; r < 8; ++r) {
        float v = (acc[r] + bs) * sc + of;
        h1s[wave][(r + 8 * hf) * 128 + och] = (half_t)fmaxf(v, 0.f);
      }
    }
    lds_fence();
    // ---- layer 2: 128 -> 64, bn, relu ----
    for (int nt = 0; nt < 4; ++nt) {
      v8f acc = {};
#pragma unroll
      for (int kt = 0; kt < 4; ++kt) {
        v16h a = frag_f16(&h1s[wave][lm * 128 + kt * 32], hf);
        v16h w = frag_f32(hp.w2 + (nt * 16 + lm) * 128 + kt * 32, hf);
        acc = wmma16(a, w, acc);
      }
      int och = nt * 16 + lm;
      float sc = hp.g2[och] * bnscale, of = hp.bb2[och], bs = hp.b2[och];
#pragma unroll
      for (int r = 0; r < 8; ++r) {
        float v = (acc[r] + bs) * sc + of;
        h2s[wave][(r + 8 * hf) * 64 + och] = (half_t)fmaxf(v, 0.f);
      }
    }
    lds_fence();
    // ---- layer 3: 64 -> dout (padded to 16) ----
    {
      int dout = (hi == 0) ? 15 : 3;
      int wrow = imin(lm, dout - 1);       // clamped weight row
      v8f acc = {};
#pragma unroll
      for (int kt = 0; kt < 2; ++kt) {
        v16h a = frag_f16(&h2s[wave][lm * 64 + kt * 32], hf);
        v16h w = frag_f32(hp.w3 + wrow * 64 + kt * 32, hf);
        if (lm >= dout) w = v16h{};        // zero padded output columns
        acc = wmma16(a, w, acc);
      }
      float bs = (lm < dout) ? hp.b3[lm] : 0.f;
#pragma unroll
      for (int r = 0; r < 8; ++r)
        outs[wave][hi][(r + 8 * hf) * 16 + lm] = acc[r] + bs;
    }
    lds_fence();
  }

  // ---- epilogue: softmax(seg), mask, pose pooling ----
  if (lane < 16) {
    int row = row0 + lane;
    if (row < N) {
      int bid = 0;
      for (int i = 0; i < NB; ++i) bid += (row >= offset[i]) ? 1 : 0;
      bid = imin(bid, NB - 1);
      float lg[15], sm[15];
      float mx = -3.4e38f;
      for (int c = 0; c < 15; ++c) {
        lg[c] = outs[wave][0][lane * 16 + c];
        seg_out[(size_t)row * 15 + c] = lg[c];
        mx = fmaxf(mx, lg[c]);
      }
      float ssum = 0.f;
      for (int c = 0; c < 15; ++c) { sm[c] = __expf(lg[c] - mx); ssum += sm[c]; }
      float inv = 1.f / ssum, maxp = 0.f;
      for (int c = 0; c < 15; ++c) { sm[c] *= inv; maxp = fmaxf(maxp, sm[c]); }
      bool kmask = (maxp > 0.6f) && (sm[0] < 0.6f);
      if (kmask) {
        float pr[3];
        for (int j = 0; j < 3; ++j)
          pr[j] = outs[wave][1][lane * 16 + j] + coord[(size_t)row * 3 + j];
        for (int c = 0; c < 15; ++c) {
          float p = sm[c];
          atomicAdd(&den[bid * 15 + c], p);
          for (int j = 0; j < 3; ++j)
            atomicAdd(&num[(bid * 15 + c) * 3 + j], pr[j] * p);
        }
      }
    }
  }
}

// ---------------------------------------------------------------------------
// 3. pose_r = num/(den+eps); emit pose_r_k (classes 1..14)
// ---------------------------------------------------------------------------
__global__ void pose_r_kernel(const float* num, const float* den,
                              float* out_posek, float* posek_ws) {
  int i = blockIdx.x * blockDim.x + threadIdx.x;
  if (i >= NB * 15) return;
  int b = i / 15, cls = i % 15;
  float inv = 1.f / (den[i] + 1e-8f);
  if (cls >= 1) {
    int k = b * KPT + (cls - 1);
    for (int j = 0; j < 3; ++j) {
      float v = num[i * 3 + j] * inv;
      out_posek[k * 3 + j] = v;
      posek_ws[k * 3 + j] = v;
    }
  }
}

// ---------------------------------------------------------------------------
// 4. token embedding: build x0 = kfc_tr([q|mem]) for all valid tokens
// ---------------------------------------------------------------------------
__global__ __launch_bounds__(64) void token_kernel(
    const float* feat, const float* coord, const int* offset,
    const float* posek, const float* kpt_emb, const float* kfc_w,
    const float* kfc_b, const float* kfc_tr_w, const float* kfc_tr_b,
    float* x) {
  int b = blockIdx.y, t = blockIdx.x, c = threadIdx.x;
  int st = seg_start(offset, b);
  int Sb = (offset[b] - st) + KPT;
  if (t >= Sb) return;
  float o = kfc_tr_b[c];
  if (t < KPT) {
    __shared__ float qv[64];
    __shared__ float p3[3];
    const float* pk = posek + (b * KPT + t) * 3;
    float a = kfc_b[c];
    for (int i = 0; i < 3; ++i) a += kfc_w[c * 67 + i] * pk[i];
    for (int i = 0; i < 64; ++i) a += kfc_w[c * 67 + 3 + i] * kpt_emb[t * 64 + i];
    qv[c] = a;
    if (c < 3) p3[c] = pk[c];
    __syncthreads();
    for (int i = 0; i < 3; ++i) o += kfc_tr_w[c * 67 + i] * p3[i];
    for (int i = 0; i < 64; ++i) o += kfc_tr_w[c * 67 + 3 + i] * qv[i];
  } else {
    int p = st + (t - KPT);
    for (int i = 0; i < 3; ++i)
      o += kfc_tr_w[c * 67 + i] * coord[(size_t)p * 3 + i];
    for (int i = 0; i < 64; ++i)
      o += kfc_tr_w[c * 67 + 3 + i] * feat[(size_t)p * 64 + i];
  }
  x[((size_t)b * SEQ + t) * 64 + c] = o;
}

// ---------------------------------------------------------------------------
// 5a. QKV projection (64 -> 192), q pre-scaled by 1/sqrt(hd), stored f16
//     layout: q/k/v[( (b*4+h)*SEQ + t )*16 + d]
// ---------------------------------------------------------------------------
__global__ __launch_bounds__(256) void qkv_kernel(
    const float* x, const float* in_w, const float* in_b, half_t* qh,
    half_t* kh, half_t* vh, const int* offset) {
  const int b = blockIdx.y;
  const int Sb = (offset[b] - seg_start(offset, b)) + KPT;
  const int wave = threadIdx.x >> 5, lane = threadIdx.x & 31;
  const int lm = lane & 15, hf = lane >> 4;
  const int t0 = (blockIdx.x * 8 + wave) * 16;
  if (t0 >= Sb) return;
  const float* X = x + (size_t)b * SEQ * 64;
  const float* arow = X + (size_t)imin(t0 + lm, Sb - 1) * 64;

  for (int nt = 0; nt < 12; ++nt) {
    v8f acc = {};
#pragma unroll
    for (int kt = 0; kt < 2; ++kt) {
      v16h a = frag_f32(arow + kt * 32, hf);
      v16h w = frag_f32(in_w + (nt * 16 + lm) * 64 + kt * 32, hf);
      acc = wmma16(a, w, acc);
    }
    // q/k/v selection and head index are uniform per n-tile
    half_t* dst = (nt < 4) ? qh : (nt < 8) ? kh : vh;
    const float scale = (nt < 4) ? 0.25f : 1.f;
    dst += (size_t)(b * 4 + (nt & 3)) * SEQ * 16;
    float bs = in_b[nt * 16 + lm];
#pragma unroll
    for (int r = 0; r < 8; ++r) {
      int t = t0 + r + 8 * hf;
      if (t < Sb) dst[(size_t)t * 16 + lm] = (half_t)((acc[r] + bs) * scale);
    }
  }
}

// ---------------------------------------------------------------------------
// 5b. flash attention: one wave per (b,h,qtile); online softmax over 32-key
//     chunks; QK^T and P*V both WMMA; P staged via LDS + s_wait_dscnt
// ---------------------------------------------------------------------------
__global__ __launch_bounds__(32) void attn_kernel(const half_t* qh,
                                                  const half_t* kh,
                                                  const half_t* vh,
                                                  float* attn_out,
                                                  const int* offset) {
  __shared__ half_t pls[16 * 32];
  const int bh = blockIdx.y, b = bh >> 2, h = bh & 3;
  const int Sb = (offset[b] - seg_start(offset, b)) + KPT;
  const int q0 = blockIdx.x * 16;
  if (q0 >= Sb) return;
  const int lane = threadIdx.x & 31, lm = lane & 15, hf = lane >> 4;
  const half_t* Q = qh + (size_t)bh * SEQ * 16;
  const half_t* Kp = kh + (size_t)bh * SEQ * 16;
  const half_t* Vp = vh + (size_t)bh * SEQ * 16;

  v16h a_q = frag_f16_pad16(Q + (size_t)imin(q0 + lm, Sb - 1) * 16, hf);

  v8f acc = {};
  float m[8], l[8];
#pragma unroll
  for (int r = 0; r < 8; ++r) { m[r] = -3.0e38f; l[r] = 0.f; }

  const int nck = (Sb + 31) / 32;
  for (int c = 0; c < nck; ++c) {
    const int key0 = c * 32;
    // clamped key rows: always read valid finite data; masked in softmax
    v16h b_k0 =
        frag_f16_pad16(Kp + (size_t)imin(key0 + lm, Sb - 1) * 16, hf);
    v16h b_k1 =
        frag_f16_pad16(Kp + (size_t)imin(key0 + 16 + lm, Sb - 1) * 16, hf);
    v8f z0 = {}, z1 = {};
    v8f s0 = wmma16(a_q, b_k0, z0);
    v8f s1 = wmma16(a_q, b_k1, z1);

    bool v0 = (key0 + lm) < Sb, v1 = (key0 + 16 + lm) < Sb;
    float p0[8], p1[8];
#pragma unroll
    for (int r = 0; r < 8; ++r) {
      float c0 = v0 ? s0[r] : -3.0e38f;
      float c1 = v1 ? s1[r] : -3.0e38f;
      float mx = fmaxf(c0, c1);
      for (int d = 1; d < 16; d <<= 1) mx = fmaxf(mx, __shfl_xor(mx, d));
      float nm = fmaxf(m[r], mx);
      float sc = __expf(m[r] - nm);
      p0[r] = v0 ? __expf(s0[r] - nm) : 0.f;
      p1[r] = v1 ? __expf(s1[r] - nm) : 0.f;
      float rs = p0[r] + p1[r];
      for (int d = 1; d < 16; d <<= 1) rs += __shfl_xor(rs, d);
      l[r] = l[r] * sc + rs;
      acc[r] *= sc;
      m[r] = nm;
    }
#pragma unroll
    for (int r = 0; r < 8; ++r) {
      int mm = r + 8 * hf;
      pls[mm * 32 + lm] = (half_t)p0[r];
      pls[mm * 32 + 16 + lm] = (half_t)p1[r];
    }
    lds_fence();
    v16h a_p = frag_f16(&pls[lm * 32], hf);
    // V gather: per-element clamped key (p==0 kills clamped-garbage terms)
    v16h b_v;
#pragma unroll
    for (int e = 0; e < 16; ++e) {
      int kl = (e < 8) ? (8 * hf + e) : (16 + 8 * hf + (e - 8));
      int key = imin(key0 + kl, Sb - 1);
      b_v[e] = Vp[(size_t)key * 16 + lm];
    }
    acc = wmma16(a_p, b_v, acc);
    lds_fence();  // keep P reads ahead of next chunk's writes
  }
#pragma unroll
  for (int r = 0; r < 8; ++r) {
    int t = q0 + r + 8 * hf;
    if (t < Sb)
      attn_out[((size_t)b * SEQ + t) * 64 + h * 16 + lm] = acc[r] / l[r];
  }
}

// ---------------------------------------------------------------------------
// 5c. out-proj (64x64) + residual + LayerNorm(n1)
// ---------------------------------------------------------------------------
__global__ __launch_bounds__(256) void proj_ln_kernel(
    const float* attn, const float* out_w, const float* out_b, const float* g,
    const float* bb, float* x, const int* offset) {
  __shared__ float olds[8][16 * 64];
  const int b = blockIdx.y;
  const int Sb = (offset[b] - seg_start(offset, b)) + KPT;
  const int wave = threadIdx.x >> 5, lane = threadIdx.x & 31;
  const int lm = lane & 15, hf = lane >> 4;
  const int t0 = (blockIdx.x * 8 + wave) * 16;
  if (t0 >= Sb) return;
  const float* arow =
      attn + (size_t)b * SEQ * 64 + (size_t)imin(t0 + lm, Sb - 1) * 64;

  for (int nt = 0; nt < 4; ++nt) {
    v8f acc = {};
#pragma unroll
    for (int kt = 0; kt < 2; ++kt) {
      v16h a = frag_f32(arow + kt * 32, hf);
      v16h w = frag_f32(out_w + (nt * 16 + lm) * 64 + kt * 32, hf);
      acc = wmma16(a, w, acc);
    }
    int och = nt * 16 + lm;
    float bs = out_b[och];
#pragma unroll
    for (int r = 0; r < 8; ++r)
      olds[wave][(r + 8 * hf) * 64 + och] = acc[r] + bs;
  }
  lds_fence();
  if (lane < 16) {
    int t = t0 + lane;
    if (t < Sb)
      ln_row64(x + ((size_t)b * SEQ + t) * 64, &olds[wave][lane * 64], g, bb);
  }
}

// ---------------------------------------------------------------------------
// 5d. FFN 64->256 relu ->64 + residual + LayerNorm(n2)
// ---------------------------------------------------------------------------
__global__ __launch_bounds__(128) void ffn_kernel(
    const float* w1, const float* b1, const float* w2, const float* b2,
    const float* g, const float* bb, float* x, const int* offset) {
  __shared__ half_t hls[4][16 * 256];
  __shared__ float  yls[4][16 * 64];
  const int b = blockIdx.y;
  const int Sb = (offset[b] - seg_start(offset, b)) + KPT;
  const int wave = threadIdx.x >> 5, lane = threadIdx.x & 31;
  const int lm = lane & 15, hf = lane >> 4;
  const int t0 = (blockIdx.x * 4 + wave) * 16;
  if (t0 >= Sb) return;
  float* X = x + (size_t)b * SEQ * 64;
  const float* arow = X + (size_t)imin(t0 + lm, Sb - 1) * 64;

  for (int nt = 0; nt < 16; ++nt) {
    v8f acc = {};
#pragma unroll
    for (int kt = 0; kt < 2; ++kt) {
      v16h a = frag_f32(arow + kt * 32, hf);
      v16h w = frag_f32(w1 + (nt * 16 + lm) * 64 + kt * 32, hf);
      acc = wmma16(a, w, acc);
    }
    int och = nt * 16 + lm;
    float bs = b1[och];
#pragma unroll
    for (int r = 0; r < 8; ++r)
      hls[wave][(r + 8 * hf) * 256 + och] = (half_t)fmaxf(acc[r] + bs, 0.f);
  }
  lds_fence();
  for (int nt = 0; nt < 4; ++nt) {
    v8f acc = {};
#pragma unroll
    for (int kt = 0; kt < 8; ++kt) {
      v16h a = frag_f16(&hls[wave][lm * 256 + kt * 32], hf);
      v16h w = frag_f32(w2 + (nt * 16 + lm) * 256 + kt * 32, hf);
      acc = wmma16(a, w, acc);
    }
    int och = nt * 16 + lm;
    float bs = b2[och];
#pragma unroll
    for (int r = 0; r < 8; ++r)
      yls[wave][(r + 8 * hf) * 64 + och] = acc[r] + bs;
  }
  lds_fence();
  if (lane < 16) {
    int t = t0 + lane;
    if (t < Sb) ln_row64(X + (size_t)t * 64, &yls[wave][lane * 64], g, bb);
  }
}

// ---------------------------------------------------------------------------
// 6. final knorm + tiny MLP on the K query tokens -> pose
// ---------------------------------------------------------------------------
__global__ __launch_bounds__(64) void final_kernel(
    const float* x, const float* kg, const float* kb, const float* cw1,
    const float* cb1, const float* cw2, const float* cb2, const float* cw3,
    const float* cb3, float* out_pose) {
  __shared__ float xn[64], h1[16], h2[8];
  int b = blockIdx.x, c = threadIdx.x;
  for (int t = 0; t < KPT; ++t) {
    const float* xr = x + ((size_t)b * SEQ + t) * 64;
    float mu = 0.f;
    for (int i = 0; i < 64; ++i) mu += xr[i];
    mu *= (1.f / 64.f);
    float var = 0.f;
    for (int i = 0; i < 64; ++i) { float d = xr[i] - mu; var += d * d; }
    var *= (1.f / 64.f);
    xn[c] = (xr[c] - mu) * rsqrtf(var + 1e-5f) * kg[c] + kb[c];
    __syncthreads();
    if (c < 16) {
      float a = cb1[c];
      for (int i = 0; i < 64; ++i) a += cw1[c * 64 + i] * xn[i];
      h1[c] = fmaxf(a, 0.f);
    }
    __syncthreads();
    if (c < 8) {
      float a = cb2[c];
      for (int i = 0; i < 16; ++i) a += cw2[c * 16 + i] * h1[i];
      h2[c] = a;
    }
    __syncthreads();
    if (c < 3) {
      float a = cb3[c];
      for (int i = 0; i < 8; ++i) a += cw3[c * 8 + i] * h2[i];
      out_pose[(b * KPT + t) * 3 + c] = a;
    }
    __syncthreads();
  }
}

// ---------------------------------------------------------------------------
// host launcher
// ---------------------------------------------------------------------------
extern "C" void kernel_launch(void* const* d_in, const int* in_sizes, int n_in,
                              void* d_out, int out_size, void* d_ws,
                              size_t ws_size, hipStream_t stream) {
  const float* feat  = (const float*)d_in[0];
  const float* coord = (const float*)d_in[1];
  const int*   offset = (const int*)d_in[2];
  const int N = in_sizes[0] / CDIM;

  // params in jax pytree (sorted dict-key) flatten order
  const float* cb1 = (const float*)d_in[3];
  const float* cb2 = (const float*)d_in[4];
  const float* cb3 = (const float*)d_in[5];
  const float* cw1 = (const float*)d_in[6];
  const float* cw2 = (const float*)d_in[7];
  const float* cw3 = (const float*)d_in[8];
  const float* kfc_b = (const float*)d_in[9];
  const float* kfc_tr_b = (const float*)d_in[10];
  const float* kfc_tr_w = (const float*)d_in[11];
  const float* kfc_w = (const float*)d_in[12];
  const float* knorm_b = (const float*)d_in[13];
  const float* knorm_g = (const float*)d_in[14];
  const float* kpt_emb = (const float*)d_in[15];
  struct Layer {
    const float *in_b, *in_w, *lin1_b, *lin1_w, *lin2_b, *lin2_w;
    const float *n1_b, *n1_g, *n2_b, *n2_g, *out_b, *out_w;
  } L[3];
  for (int l = 0; l < 3; ++l) {
    int s = 16 + 12 * l;
    L[l].in_b = (const float*)d_in[s + 0];
    L[l].in_w = (const float*)d_in[s + 1];
    L[l].lin1_b = (const float*)d_in[s + 2];
    L[l].lin1_w = (const float*)d_in[s + 3];
    L[l].lin2_b = (const float*)d_in[s + 4];
    L[l].lin2_w = (const float*)d_in[s + 5];
    L[l].n1_b = (const float*)d_in[s + 6];
    L[l].n1_g = (const float*)d_in[s + 7];
    L[l].n2_b = (const float*)d_in[s + 8];
    L[l].n2_g = (const float*)d_in[s + 9];
    L[l].out_b = (const float*)d_in[s + 10];
    L[l].out_w = (const float*)d_in[s + 11];
  }
  auto mk_head = [&](int s) {
    HeadParams h;
    h.b1 = (const float*)d_in[s + 0];
    h.b2 = (const float*)d_in[s + 1];
    h.b3 = (const float*)d_in[s + 2];
    h.bb1 = (const float*)d_in[s + 3];
    h.bb2 = (const float*)d_in[s + 4];
    h.g1 = (const float*)d_in[s + 5];
    h.g2 = (const float*)d_in[s + 6];
    h.w1 = (const float*)d_in[s + 7];
    h.w2 = (const float*)d_in[s + 8];
    h.w3 = (const float*)d_in[s + 9];
    return h;
  };
  HeadParams regp = mk_head(52);
  HeadParams segp = mk_head(62);

  // workspace carve-up
  char* W = (char*)d_ws;
  size_t off = 0;
  auto alloc = [&](size_t bytes) -> void* {
    void* p = W + off;
    off += (bytes + 255) & ~(size_t)255;
    return p;
  };
  float* fmax = (float*)alloc(NB * 64 * 4);
  float* fmean = (float*)alloc(NB * 64 * 4);
  float* num = (float*)alloc(NB * 15 * 3 * 4);
  float* den = (float*)alloc(NB * 15 * 4);
  float* posek = (float*)alloc(NB * KPT * 3 * 4);
  float* x = (float*)alloc((size_t)NB * SEQ * 64 * 4);
  half_t* qh = (half_t*)alloc((size_t)NB * 4 * SEQ * 16 * 2);
  half_t* kh = (half_t*)alloc((size_t)NB * 4 * SEQ * 16 * 2);
  half_t* vh = (half_t*)alloc((size_t)NB * 4 * SEQ * 16 * 2);
  float* attn = (float*)alloc((size_t)NB * SEQ * 64 * 4);

  float* out = (float*)d_out;
  float* out_seg = out;
  float* out_posek = out + (size_t)N * 15;
  float* out_pose = out_posek + NB * KPT * 3;

  init_kernel<<<4, 256, 0, stream>>>(num, den);
  pool_kernel<<<NB, 64, 0, stream>>>(feat, offset, fmax, fmean);
  heads_kernel<<<(N + 63) / 64, 128, 0, stream>>>(
      feat, coord, offset, fmax, fmean, segp, regp, out_seg, num, den, N);
  pose_r_kernel<<<1, 256, 0, stream>>>(num, den, out_posek, posek);
  token_kernel<<<dim3(SEQ, NB), 64, 0, stream>>>(
      feat, coord, offset, posek, kpt_emb, kfc_w, kfc_b, kfc_tr_w, kfc_tr_b, x);

  const int gq = (SEQ + 127) / 128;   // 8 waves x 16 tokens
  const int gf = (SEQ + 63) / 64;     // 4 waves x 16 tokens
  const int ga = (SEQ + 15) / 16;     // 1 wave per q-tile
  for (int l = 0; l < 3; ++l) {
    qkv_kernel<<<dim3(gq, NB), 256, 0, stream>>>(x, L[l].in_w, L[l].in_b, qh,
                                                 kh, vh, offset);
    attn_kernel<<<dim3(ga, NB * 4), 32, 0, stream>>>(qh, kh, vh, attn, offset);
    proj_ln_kernel<<<dim3(gq, NB), 256, 0, stream>>>(
        attn, L[l].out_w, L[l].out_b, L[l].n1_g, L[l].n1_b, x, offset);
    ffn_kernel<<<dim3(gf, NB), 128, 0, stream>>>(
        L[l].lin1_w, L[l].lin1_b, L[l].lin2_w, L[l].lin2_b, L[l].n2_g,
        L[l].n2_b, x, offset);
  }
  final_kernel<<<NB, 64, 0, stream>>>(x, knorm_g, knorm_b, cw1, cb1, cw2, cb2,
                                      cw3, cb3, out_pose);
}
